// Block_57569741635983
// MI455X (gfx1250) — matrix operand order repaired and checked by reading, not compiled, not run on hardware
//
#include <hip/hip_runtime.h>
#include <hip/hip_bf16.h>

// ---------------------------------------------------------------------------
// GPT block on gfx1250: bf16 WMMA GEMMs (f32 accum) + flash attention.
// Round 3: DPP row_xmask via template-constant ctrl (fixes ICE requirement),
//          128x128 GEMM tiles, software-pipelined global->LDS staging.
// ---------------------------------------------------------------------------

#define BB 4
#define TT 2048
#define CC 1024
#define HH 16
#define HS 64
#define MM (BB*TT)          // 8192 rows

typedef __attribute__((ext_vector_type(16))) __bf16 v16bf;
typedef __attribute__((ext_vector_type(8)))  float  v8f;

__device__ __forceinline__ unsigned short f2bf(float f) {
    unsigned u = __builtin_bit_cast(unsigned, f);
    unsigned r = u + 0x7FFFu + ((u >> 16) & 1u);   // round-to-nearest-even
    return (unsigned short)(r >> 16);
}

__device__ __forceinline__ v16bf load_frag(const unsigned short* p0,
                                           const unsigned short* p1) {
    union { uint4 q[2]; v16bf f; } u;
    u.q[0] = *(const uint4*)p0;
    u.q[1] = *(const uint4*)p1;
    return u.f;
}

__device__ __forceinline__ v8f wmma_bf16(v16bf a, v16bf b, v8f c) {
    return __builtin_amdgcn_wmma_f32_16x16x32_bf16(
        /*neg_a=*/false, a, /*neg_b=*/false, b,
        /*c_mod=*/(short)0, c, /*reuse_a=*/false, /*reuse_b=*/false);
}

// 16-lane butterfly step via DPP ROW_XMASK (0x160 | xor_mask): single VALU op,
// stays within each 16-lane row half -- matches the WMMA C-layout row halves.
// DPP ctrl must be an integer constant expression -> template parameter.
template <int MASK>
__device__ __forceinline__ float dpp_xor16(float v) {
    int x = __builtin_amdgcn_update_dpp(
        0, __builtin_bit_cast(int, v), 0x160 | MASK, 0xf, 0xf, true);
    return __builtin_bit_cast(float, x);
}
__device__ __forceinline__ float red16_max(float v) {
    v = fmaxf(v, dpp_xor16<1>(v));
    v = fmaxf(v, dpp_xor16<2>(v));
    v = fmaxf(v, dpp_xor16<4>(v));
    v = fmaxf(v, dpp_xor16<8>(v));
    return v;
}
__device__ __forceinline__ float red16_add(float v) {
    v += dpp_xor16<1>(v);
    v += dpp_xor16<2>(v);
    v += dpp_xor16<4>(v);
    v += dpp_xor16<8>(v);
    return v;
}

// ---------------------------------------------------------------------------
// f32 -> bf16 conversion (weights)
// ---------------------------------------------------------------------------
__global__ __launch_bounds__(256) void cvt_bf16_kernel(const float* __restrict__ in,
                                                       unsigned short* __restrict__ out,
                                                       size_t n) {
    size_t i = (size_t)blockIdx.x * 256 + threadIdx.x;
    if (i < n) out[i] = f2bf(in[i]);
}

// ---------------------------------------------------------------------------
// LayerNorm: one block per row of C=1024, f32 in -> bf16 out
// ---------------------------------------------------------------------------
__global__ __launch_bounds__(256) void ln_kernel(const float* __restrict__ x,
                                                 const float* __restrict__ g,
                                                 const float* __restrict__ b,
                                                 unsigned short* __restrict__ out) {
    const int row = blockIdx.x;
    const float* xr = x + (size_t)row * CC;
    __shared__ float red[256];

    float s = 0.f;
    for (int i = threadIdx.x; i < CC; i += 256) s += xr[i];
    red[threadIdx.x] = s; __syncthreads();
    for (int o = 128; o > 0; o >>= 1) {
        if (threadIdx.x < o) red[threadIdx.x] += red[threadIdx.x + o];
        __syncthreads();
    }
    float mu = red[0] * (1.0f / CC);
    __syncthreads();

    float v = 0.f;
    for (int i = threadIdx.x; i < CC; i += 256) { float d = xr[i] - mu; v += d * d; }
    red[threadIdx.x] = v; __syncthreads();
    for (int o = 128; o > 0; o >>= 1) {
        if (threadIdx.x < o) red[threadIdx.x] += red[threadIdx.x + o];
        __syncthreads();
    }
    float rstd = rsqrtf(red[0] * (1.0f / CC) + 1e-5f);

    for (int i = threadIdx.x; i < CC; i += 256)
        out[(size_t)row * CC + i] = f2bf((xr[i] - mu) * rstd * g[i] + b[i]);
}

// ---------------------------------------------------------------------------
// Tiled bf16 WMMA GEMM:  out[M,N] = A[M,K] @ W[K,N] + bias  (+ epilogue)
// Block tile 128x128, 8 waves (4x2), each wave 32x64 (2x4 WMMA tiles),
// K-step 32, software-pipelined global->LDS staging.
// EPI: 0 = bias, store bf16
//      1 = bias + residual(f32), store f32
//      2 = bias + gelu(tanh), store bf16
// ---------------------------------------------------------------------------
template <int EPI>
__global__ __launch_bounds__(256) void gemm_bf16_kernel(
    const unsigned short* __restrict__ A,
    const unsigned short* __restrict__ W,
    const float* __restrict__ bias,
    const float* __restrict__ resid,
    float* __restrict__ outF,
    unsigned short* __restrict__ outB,
    int M, int N, int K)
{
    const int t    = threadIdx.x;
    const int lane = t & 31;
    const int wave = t >> 5;
    const int wm   = wave & 3;        // wave row (0..3) -> 32 M-rows each
    const int wn   = wave >> 2;       // wave col (0..1) -> 64 N-cols each
    const int l16  = lane & 15;
    const int hh   = lane >> 4;       // lane half (0/1)
    const int bm   = blockIdx.y * 128;
    const int bn   = blockIdx.x * 128;

    __shared__ alignas(16) unsigned short As[128 * 32];   // [m][k]
    __shared__ alignas(16) unsigned short Bs[128 * 32];   // [n][k] (transposed)

    v8f acc[2][4] = {};
    uint4 ra[2], rb[2];

    // A chunk i: c = t*2+i (512 chunks of 8): row = c/4, kc = (c%4)*8
    // B chunk i: c = t*2+i (512 chunks of 8): kr  = c/16, nc = (c%16)*8
    auto load_tiles = [&](int k0) {
        #pragma unroll
        for (int i = 0; i < 2; i++) {
            int c = t * 2 + i;
            int arow = c >> 2, akc = (c & 3) * 8;
            ra[i] = *(const uint4*)(A + (size_t)(bm + arow) * K + k0 + akc);
            int bkr = c >> 4, bnc = (c & 15) * 8;
            rb[i] = *(const uint4*)(W + (size_t)(k0 + bkr) * N + bn + bnc);
        }
    };
    auto store_tiles = [&]() {
        #pragma unroll
        for (int i = 0; i < 2; i++) {
            int c = t * 2 + i;
            int arow = c >> 2, akc = (c & 3) * 8;
            *(uint4*)&As[arow * 32 + akc] = ra[i];
            int bkr = c >> 4, bnc = (c & 15) * 8;
            unsigned short tmp[8]; *(uint4*)tmp = rb[i];
            #pragma unroll
            for (int j = 0; j < 8; j++) Bs[(bnc + j) * 32 + bkr] = tmp[j];
        }
    };

    load_tiles(0);
    for (int k0 = 0; k0 < K; k0 += 32) {
        __syncthreads();               // previous compute done, LDS reusable
        store_tiles();
        __syncthreads();
        if (k0 + 32 < K) load_tiles(k0 + 32);   // overlap next fetch w/ WMMAs

        #pragma unroll
        for (int mt = 0; mt < 2; mt++) {
            const int arow = wm * 32 + mt * 16 + l16;
            v16bf af = load_frag(&As[arow * 32 + hh * 8],
                                 &As[arow * 32 + 16 + hh * 8]);
            #pragma unroll
            for (int nt = 0; nt < 4; nt++) {
                const int bcol = wn * 64 + nt * 16 + l16;
                const unsigned short* bp = &Bs[bcol * 32 + hh * 8];
                v16bf bf = load_frag(bp, bp + 16);
                acc[mt][nt] = wmma_bf16(af, bf, acc[mt][nt]);
            }
        }
    }

    // --- epilogue ---
    #pragma unroll
    for (int mt = 0; mt < 2; mt++)
    #pragma unroll
    for (int nt = 0; nt < 4; nt++) {
        #pragma unroll
        for (int r = 0; r < 8; r++) {
            int row = bm + wm * 32 + mt * 16 + r + hh * 8;
            int col = bn + wn * 64 + nt * 16 + l16;
            float v = acc[mt][nt][r] + bias[col];
            size_t idx = (size_t)row * N + col;
            if (EPI == 1) {
                outF[idx] = resid[idx] + v;
            } else if (EPI == 2) {
                float tg = 0.7978845608028654f * (v + 0.044715f * v * v * v);
                v = 0.5f * v * (1.0f + tanhf(tg));
                outB[idx] = f2bf(v);
            } else {
                outB[idx] = f2bf(v);
            }
        }
    }
}

// ---------------------------------------------------------------------------
// Flash attention (causal). qkv: bf16 [B][T][3C]. out: bf16 [B][T][C].
// Block = 128 q rows, 8 waves x 16 rows. kv tiles of 32, pipelined staging.
// ---------------------------------------------------------------------------
__global__ __launch_bounds__(256) void attn_kernel(const unsigned short* __restrict__ qkv,
                                                   unsigned short* __restrict__ out)
{
    const int t    = threadIdx.x;
    const int lane = t & 31;
    const int wave = t >> 5;
    const int l16  = lane & 15;
    const int hh   = lane >> 4;
    const int bh   = blockIdx.y;
    const int b    = bh >> 4;
    const int h    = bh & 15;
    const int qbase = blockIdx.x * 128;
    const int C3   = 3 * CC;

    const size_t baseQ = (size_t)b * TT * C3 + (size_t)h * HS;
    const size_t baseK = baseQ + CC;
    const size_t baseV = baseQ + 2 * CC;

    __shared__ alignas(16) unsigned short Qs[128 * 64];   // [q][hs]
    __shared__ alignas(16) unsigned short Ks[32 * 64];    // [kv][hs]
    __shared__ alignas(16) unsigned short Vt[64 * 32];    // [hs][kv]
    __shared__ alignas(16) unsigned short Ps[8][16 * 32]; // per-wave P tile

    // --- load Q block (128x64), 4 x uint4 per thread ---
    #pragma unroll
    for (int i = 0; i < 4; i++) {
        int c   = t * 4 + i;                 // 1024 chunks of 8
        int row = c >> 3;
        int col = (c & 7) * 8;
        *(uint4*)&Qs[row * 64 + col] =
            *(const uint4*)&qkv[baseQ + (size_t)(qbase + row) * C3 + col];
    }
    __syncthreads();

    const int qrow0 = wave * 16;
    v16bf qf[2];
    {
        const int arow = qrow0 + l16;
        qf[0] = load_frag(&Qs[arow * 64 + hh * 8],      &Qs[arow * 64 + 16 + hh * 8]);
        qf[1] = load_frag(&Qs[arow * 64 + 32 + hh * 8], &Qs[arow * 64 + 48 + hh * 8]);
    }

    v8f o[4] = {};
    float m[8], l[8];
    #pragma unroll
    for (int r = 0; r < 8; r++) { m[r] = -1e30f; l[r] = 0.f; }

    // KV staging: thread t handles one 8-elem chunk of K and of V per tile
    const int krow = t >> 3;                 // 0..31
    const int kcol = (t & 7) * 8;            // 0..56
    uint4 rk, rv;
    auto load_kv = [&](int kv0) {
        rk = *(const uint4*)&qkv[baseK + (size_t)(kv0 + krow) * C3 + kcol];
        rv = *(const uint4*)&qkv[baseV + (size_t)(kv0 + krow) * C3 + kcol];
    };

    const int kvEnd = qbase + 128;
    load_kv(0);
    for (int kv0 = 0; kv0 < kvEnd; kv0 += 32) {
        __syncthreads();
        *(uint4*)&Ks[krow * 64 + kcol] = rk;
        {
            unsigned short tmp[8]; *(uint4*)tmp = rv;
            #pragma unroll
            for (int j = 0; j < 8; j++) Vt[(kcol + j) * 32 + krow] = tmp[j];
        }
        __syncthreads();
        if (kv0 + 32 < kvEnd) load_kv(kv0 + 32);

        // --- S = Q K^T for two 16-wide kv chunks ---
        v8f s[2];
        #pragma unroll
        for (int c16 = 0; c16 < 2; c16++) {
            v8f sc = {};
            #pragma unroll
            for (int ks = 0; ks < 2; ks++) {
                const int kvcol = c16 * 16 + l16;
                const unsigned short* bp = &Ks[kvcol * 64 + ks * 32 + hh * 8];
                v16bf kf = load_frag(bp, bp + 16);
                sc = wmma_bf16(qf[ks], kf, sc);
            }
            s[c16] = sc;
        }

        // --- online softmax (rows r + hh*8, cols across 16 lanes) ---
        const float scale = 0.125f;  // 1/sqrt(64)
        #pragma unroll
        for (int r = 0; r < 8; r++) {
            const int q = qbase + qrow0 + r + hh * 8;
            #pragma unroll
            for (int c16 = 0; c16 < 2; c16++) {
                int kvc = kv0 + c16 * 16 + l16;
                float v = s[c16][r] * scale;
                if (kvc > q) v = -1e30f;
                s[c16][r] = v;
            }
            float mx = red16_max(fmaxf(s[0][r], s[1][r]));
            float nm    = fmaxf(m[r], mx);
            float alpha = __expf(m[r] - nm);
            float p0 = __expf(s[0][r] - nm);
            float p1 = __expf(s[1][r] - nm);
            s[0][r] = p0; s[1][r] = p1;
            float rs = red16_add(p0 + p1);
            l[r] = l[r] * alpha + rs;
            m[r] = nm;
            #pragma unroll
            for (int nt = 0; nt < 4; nt++) o[nt][r] *= alpha;
        }

        // --- P to LDS (C-layout -> A-layout restage), wave-private ---
        #pragma unroll
        for (int r = 0; r < 8; r++) {
            int row = r + hh * 8;
            Ps[wave][row * 32 + l16]      = f2bf(s[0][r]);
            Ps[wave][row * 32 + 16 + l16] = f2bf(s[1][r]);
        }
        // (LDS ops are in-order within a wave; same-wave RAW is safe)

        // --- O += P @ V ---
        v16bf pf = load_frag(&Ps[wave][l16 * 32 + hh * 8],
                             &Ps[wave][l16 * 32 + 16 + hh * 8]);
        #pragma unroll
        for (int nt = 0; nt < 4; nt++) {
            const int hs = nt * 16 + l16;
            const unsigned short* bp = &Vt[hs * 32 + hh * 8];
            v16bf vf = load_frag(bp, bp + 16);
            o[nt] = wmma_bf16(pf, vf, o[nt]);
        }
    }

    // --- normalize and write bf16 out[b][q][h*64+hs] ---
    float rinv[8];
    #pragma unroll
    for (int r = 0; r < 8; r++) rinv[r] = 1.0f / l[r];
    #pragma unroll
    for (int nt = 0; nt < 4; nt++)
        #pragma unroll
        for (int r = 0; r < 8; r++) {
            int q  = qbase + qrow0 + r + hh * 8;
            int hs = nt * 16 + l16;
            float v = o[nt][r] * rinv[r];
            out[(size_t)b * TT * CC + (size_t)q * CC + (size_t)h * HS + hs] = f2bf(v);
        }
}

// ---------------------------------------------------------------------------
// Launcher
// ---------------------------------------------------------------------------
extern "C" void kernel_launch(void* const* d_in, const int* in_sizes, int n_in,
                              void* d_out, int out_size, void* d_ws, size_t ws_size,
                              hipStream_t stream) {
    (void)in_sizes; (void)n_in; (void)out_size; (void)ws_size;

    const float* x        = (const float*)d_in[0];
    const float* c_attn_w = (const float*)d_in[1];
    const float* c_attn_b = (const float*)d_in[2];
    const float* c_proj_w = (const float*)d_in[3];
    const float* c_proj_b = (const float*)d_in[4];
    const float* fc_w     = (const float*)d_in[5];
    const float* fc_b     = (const float*)d_in[6];
    const float* proj_w   = (const float*)d_in[7];
    const float* proj_b   = (const float*)d_in[8];
    const float* ln1_g    = (const float*)d_in[9];
    const float* ln1_b    = (const float*)d_in[10];
    const float* ln2_g    = (const float*)d_in[11];
    const float* ln2_b    = (const float*)d_in[12];
    float* out            = (float*)d_out;

    // workspace carve-up (bf16 buffers as ushort, x1 as f32)
    char* ws = (char*)d_ws;
    size_t p = 0;
    auto take = [&](size_t bytes) { char* r = ws + p; p = (p + bytes + 255) & ~(size_t)255; return r; };
    unsigned short* wAttn  = (unsigned short*)take((size_t)CC * 3 * CC * 2);
    unsigned short* wProj  = (unsigned short*)take((size_t)CC * CC * 2);
    unsigned short* wFc    = (unsigned short*)take((size_t)CC * 4 * CC * 2);
    unsigned short* wProj2 = (unsigned short*)take((size_t)4 * CC * CC * 2);
    unsigned short* ln1o   = (unsigned short*)take((size_t)MM * CC * 2);
    unsigned short* qkv    = (unsigned short*)take((size_t)MM * 3 * CC * 2);
    unsigned short* attnO  = (unsigned short*)take((size_t)MM * CC * 2);
    float*          x1     = (float*)         take((size_t)MM * CC * 4);
    unsigned short* ln2o   = (unsigned short*)take((size_t)MM * CC * 2);
    unsigned short* hbuf   = (unsigned short*)take((size_t)MM * 4 * CC * 2);

    auto cvt = [&](const float* src, unsigned short* dst, size_t n) {
        cvt_bf16_kernel<<<dim3((unsigned)((n + 255) / 256)), 256, 0, stream>>>(src, dst, n);
    };
    cvt(c_attn_w, wAttn,  (size_t)CC * 3 * CC);
    cvt(c_proj_w, wProj,  (size_t)CC * CC);
    cvt(fc_w,     wFc,    (size_t)CC * 4 * CC);
    cvt(proj_w,   wProj2, (size_t)4 * CC * CC);

    // 1) ln1(x)
    ln_kernel<<<MM, 256, 0, stream>>>(x, ln1_g, ln1_b, ln1o);

    // 2) qkv = ln1 @ c_attn_w + b   (M=8192, N=3072, K=1024)
    gemm_bf16_kernel<0><<<dim3(3 * CC / 128, MM / 128), 256, 0, stream>>>(
        ln1o, wAttn, c_attn_b, nullptr, nullptr, qkv, MM, 3 * CC, CC);

    // 3) flash attention
    attn_kernel<<<dim3(TT / 128, BB * HH), 256, 0, stream>>>(qkv, attnO);

    // 4) x1 = x + attnO @ c_proj_w + b   (N=1024, K=1024)
    gemm_bf16_kernel<1><<<dim3(CC / 128, MM / 128), 256, 0, stream>>>(
        attnO, wProj, c_proj_b, x, x1, nullptr, MM, CC, CC);

    // 5) ln2(x1)
    ln_kernel<<<MM, 256, 0, stream>>>(x1, ln2_g, ln2_b, ln2o);

    // 6) h = gelu(ln2 @ fc_w + b)   (N=4096, K=1024)
    gemm_bf16_kernel<2><<<dim3(4 * CC / 128, MM / 128), 256, 0, stream>>>(
        ln2o, wFc, fc_b, nullptr, nullptr, hbuf, MM, 4 * CC, CC);

    // 7) out = x1 + h @ proj_w + b   (N=1024, K=4096)
    gemm_bf16_kernel<1><<<dim3(CC / 128, MM / 128), 256, 0, stream>>>(
        hbuf, wProj2, proj_b, x1, out, nullptr, MM, CC, 4 * CC);
}
